// LinearQuantized_80934363726413
// MI455X (gfx1250) — 8-bit matrix operand order repaired and checked by reading, compiled, not run on hardware
//
#include <hip/hip_runtime.h>
#include <hip/hip_bf16.h>

typedef int v2i __attribute__((ext_vector_type(2)));
typedef int v4i __attribute__((ext_vector_type(4)));
typedef int v8i __attribute__((ext_vector_type(8)));

static constexpr int  D_IN  = 4096;
static constexpr int  D_OUT = 4096;
static constexpr long M_TOT = 8L * 2048;


__global__ void fq_init(unsigned* __restrict__ s) {
    if (threadIdx.x == 0) { s[0] = 0u; s[1] = 0u; }
}

__global__ void fq_absmax(const float* __restrict__ in, size_t n4,
                          unsigned* __restrict__ out) {
    size_t i      = (size_t)blockIdx.x * blockDim.x + threadIdx.x;
    size_t stride = (size_t)gridDim.x * blockDim.x;
    const float4* in4 = (const float4*)in;
    float m = 0.0f;
    for (; i < n4; i += stride) {
        float4 v = in4[i];
        m = fmaxf(m, fmaxf(fmaxf(fabsf(v.x), fabsf(v.y)),
                           fmaxf(fabsf(v.z), fabsf(v.w))));
    }
#pragma unroll
    for (int off = 16; off > 0; off >>= 1)
        m = fmaxf(m, __shfl_xor(m, off, 32));
    if ((threadIdx.x & 31) == 0)
        atomicMax(out, __float_as_uint(m));
}

__device__ __forceinline__ float fq_scale(const unsigned* __restrict__ bits) {
    return fmaxf(__uint_as_float(*bits) * (1.0f / 127.0f), 1e-8f);
}

__device__ __forceinline__ int fq_q1(float v, float inv) {
    int a = __float2int_rn(v * inv);
    a = a > 127 ? 127 : a;
    a = a < -127 ? -127 : a;
    return a & 255;
}

__global__ void fq_quant(const float* __restrict__ in, int* __restrict__ out,
                         size_t n4, const unsigned* __restrict__ amax) {
    float inv = 1.0f / fq_scale(amax);
    size_t i      = (size_t)blockIdx.x * blockDim.x + threadIdx.x;
    size_t stride = (size_t)gridDim.x * blockDim.x;
    const float4* in4 = (const float4*)in;
    for (; i < n4; i += stride) {
        float4 v = in4[i];
        int p = fq_q1(v.x, inv) | (fq_q1(v.y, inv) << 8) |
                (fq_q1(v.z, inv) << 16) | (fq_q1(v.w, inv) << 24);
        out[i] = p;
    }
}

__global__ __launch_bounds__(256)
void fq_gemm_i8(const signed char* __restrict__ Aq,
                const signed char* __restrict__ Bq,
                const float* __restrict__ bias,
                float* __restrict__ out,
                const unsigned* __restrict__ axb,
                const unsigned* __restrict__ awb) {
    const int  K    = D_IN;
    const int  N    = D_OUT;
    const int  wid  = threadIdx.x >> 5;
    const int  lane = threadIdx.x & 31;
    const int  half = lane >> 4;
    const int  l16  = lane & 15;

    const long m0 = (long)blockIdx.y * 128 + (wid & 3) * 32;
    const long n0 = (long)blockIdx.x * 128 + (wid >> 2) * 64;

    v8i acc[2][4] = {};

    const signed char* arow0 = Aq + (m0 + l16) * (long)K;
    const signed char* arow1 = arow0 + 16L * K;

    for (int k = 0; k < K; k += 64) {
        v8i a[2];
#pragma unroll
        for (int t = 0; t < 2; ++t) {
            const signed char* ap = (t ? arow1 : arow0) + k + half * 16;
            v2i a0 = *(const v2i*)(ap + 0);
            v2i a1 = *(const v2i*)(ap + 8);
            v2i a2 = *(const v2i*)(ap + 32);
            v2i a3 = *(const v2i*)(ap + 40);
            a[t] = (v8i){a0.x, a0.y, a1.x, a1.y, a2.x, a2.y, a3.x, a3.y};
        }

#pragma unroll
        for (int j = 0; j < 4; ++j) {
            const signed char* bp =
                Bq + (n0 + j * 16 + l16) * (long)K + k + half * 16;
            v4i b0 = *(const v4i*)(bp + 0);
            v4i b1 = *(const v4i*)(bp + 32);
            v8i b = {b0.x, b0.y, b0.z, b0.w, b1.x, b1.y, b1.z, b1.w};

#pragma unroll
            for (int t = 0; t < 2; ++t) {
                acc[t][j] = __builtin_amdgcn_wmma_i32_16x16x64_iu8(
                     true, a[t],  true, b, acc[t][j],
                     false,  false);
            }
        }
    }

    const float sc = fq_scale(axb) * fq_scale(awb);

#pragma unroll
    for (int j = 0; j < 4; ++j) {
        const long n  = n0 + j * 16 + l16;
        const float bz = bias[n];
#pragma unroll
        for (int t = 0; t < 2; ++t) {
#pragma unroll
            for (int r = 0; r < 8; ++r) {
                const long m = m0 + t * 16 + r + half * 8;
                out[m * (long)N + n] = (float)acc[t][j][r] * sc + bz;
            }
        }
    }
}

extern "C" void kernel_launch(void* const* d_in, const int* in_sizes, int n_in,
                              void* d_out, int out_size, void* d_ws, size_t ws_size,
                              hipStream_t stream) {
    const float* x    = (const float*)d_in[0];
    const float* w    = (const float*)d_in[1];
    const float* bias = (const float*)d_in[2];
    float* out        = (float*)d_out;

    unsigned*    stats = (unsigned*)d_ws;
    signed char* xq    = (signed char*)d_ws + 256;
    signed char* wq    = xq + (size_t)M_TOT * D_IN;

    const size_t nx4 = (size_t)M_TOT * D_IN / 4;
    const size_t nw4 = (size_t)D_OUT * D_IN / 4;

    fq_init<<<1, 32, 0, stream>>>(stats);
    fq_absmax<<<1024, 256, 0, stream>>>(x, nx4, stats + 0);
    fq_absmax<<<512,  256, 0, stream>>>(w, nw4, stats + 1);
    fq_quant<<<2048, 256, 0, stream>>>(x, (int*)xq, nx4, stats + 0);
    fq_quant<<<1024, 256, 0, stream>>>(w, (int*)wq, nw4, stats + 1);

    dim3 grid(D_OUT / 128, (unsigned)(M_TOT / 128));
    fq_gemm_i8<<<grid, 256, 0, stream>>>(xq, wq, bias, out, stats + 0, stats + 1);
}
